// GCN_27230092657223
// MI455X (gfx1250) — compile-verified, matrix operand order checked by reading, not run on hardware
//
#include <hip/hip_runtime.h>
#include <hip/hip_bf16.h>
#include <math.h>

// MI455X / gfx1250, wave32. Main GEMMs use v_wmma_f32_16x16x32_bf16.

typedef __attribute__((ext_vector_type(16))) __bf16 v16bf;
typedef __attribute__((ext_vector_type(8)))  float  v8f;

#define NV_    4096
#define NE_    8192
#define NFV_   128
#define NFE_   64
#define NHID_  128
#define NCLS_  8

// ---- fragment helpers -------------------------------------------------------
// 16-bit A/B fragment (16x32): lane ln=lane&15 is the row (A) / col (B),
// lane half lh=lane>>4 selects K groups. Two 16B loads: K[lh*8 .. +8) and
// K[16+lh*8 .. +8). Caller passes p = base + row*ld + lh*8 (+k0).
static __device__ inline v16bf frag_ld(const __bf16* p) {
  v16bf a;
  *(uint4*)&a       = *(const uint4*)(p);
  *((uint4*)&a + 1) = *(const uint4*)(p + 16);
  return a;
}

static __device__ inline v8f wmma_bf16(v16bf a, v16bf b, v8f c) {
  return __builtin_amdgcn_wmma_f32_16x16x32_bf16(
      /*neg_a=*/false, a, /*neg_b=*/false, b,
      /*c_mod=*/(short)0, c, /*reuse_a=*/false, /*reuse_b=*/false);
}

// ---- prep kernels -----------------------------------------------------------

// d1[e] = Z[e,:] . p1   (raw Z)
__global__ __launch_bounds__(256) void k_d1(const float* __restrict__ Z,
                                            const float* __restrict__ p1,
                                            float* __restrict__ d1) {
  int e = blockIdx.x * 256 + threadIdx.x;
  float s = 0.f;
  for (int k = 0; k < NFE_; ++k) s += Z[e * NFE_ + k] * p1[k];
  d1[e] = s;
}

// bufA = bf16(T * d1[k]) (stage-1 A side), bufB = bf16(T) (stage-1 B side)
__global__ __launch_bounds__(256) void k_prep(const float* __restrict__ T,
                                              const float* __restrict__ d1,
                                              __bf16* __restrict__ bufA,
                                              __bf16* __restrict__ bufB) {
  int idx = blockIdx.x * 256 + threadIdx.x;   // NV_*NE_ = 33.5M < 2^31
  float t = T[idx];
  bufB[idx] = (__bf16)t;
  bufA[idx] = (__bf16)(t * d1[idx & (NE_ - 1)]);
}

// HvWT[f, i] = bf16( (X @ W1)[i, f] )   (transposed so WMMA B-frag loads row-major)
__global__ __launch_bounds__(256) void k_hvwt(const float* __restrict__ X,
                                              const float* __restrict__ W1,
                                              __bf16* __restrict__ HvWT) {
  int idx = blockIdx.x * 256 + threadIdx.x;   // NHID_*NV_
  int i = idx & (NV_ - 1);
  int f = idx >> 12;
  float s = 0.f;
  for (int k = 0; k < NFV_; ++k) s += X[i * NFV_ + k] * W1[k * NHID_ + f];
  HvWT[idx] = (__bf16)s;
}

// HeW2T[c, e] = bf16( (relu(Z) @ W2)[e, c] ), rows 8..15 zero-padded
__global__ __launch_bounds__(256) void k_hew2t(const float* __restrict__ Z,
                                               const float* __restrict__ W2,
                                               __bf16* __restrict__ HeW2T) {
  int idx = blockIdx.x * 256 + threadIdx.x;   // 16*NE_
  int e = idx & (NE_ - 1);
  int c = idx >> 13;
  float s = 0.f;
  if (c < NCLS_) {
    for (int k = 0; k < NFE_; ++k) {
      float z = Z[e * NFE_ + k];
      z = z > 0.f ? z : 0.f;
      s += z * W2[k * NCLS_ + c];
    }
  }
  HeW2T[idx] = (__bf16)s;
}

// ---- stage 1: Xn += (S_v .* adj_v, diag fixed) @ HvW ------------------------
// Per wave: one 16-row strip of S_v (O accumulator is 16x128 = 64 VGPRs/lane,
// so no multi-strip blocking here), looping j in this block's chunk.
__global__ __launch_bounds__(256) void k_stage1(
    const __bf16* __restrict__ Ts,   // [NV, NE]  T*d1 (bf16)
    const __bf16* __restrict__ Tr,   // [NV, NE]  T    (bf16)
    const __bf16* __restrict__ Vt,   // [NHID, NV] (X@W1)^T (bf16)
    const float*  __restrict__ adj,  // [NV, NV]
    float* __restrict__ Xn)          // [NV, NHID] (atomic accumulate)
{
  __shared__ __align__(16) __bf16 pst[8][16 * 40];
  const int lane = threadIdx.x & 31;
  const int wave = threadIdx.x >> 5;
  const int ln = lane & 15;
  const int lh = lane >> 4;
  const int i0 = (blockIdx.x * 8 + wave) * 16;
  const int jb = blockIdx.y * (NV_ / 8);
  __bf16* pl = pst[wave];

  v8f o[8];
#pragma unroll
  for (int t = 0; t < 8; ++t) o[t] = (v8f){0, 0, 0, 0, 0, 0, 0, 0};

  const __bf16* pa = Ts + (size_t)(i0 + ln) * NE_ + lh * 8;

  for (int j0 = jb; j0 < jb + NV_ / 8; j0 += 32) {
    v8f s0 = (v8f){0, 0, 0, 0, 0, 0, 0, 0};
    v8f s1 = (v8f){0, 0, 0, 0, 0, 0, 0, 0};
    const __bf16* pb0 = Tr + (size_t)(j0 + ln) * NE_ + lh * 8;
    const __bf16* pb1 = Tr + (size_t)(j0 + 16 + ln) * NE_ + lh * 8;
    for (int k0 = 0; k0 < NE_; k0 += 32) {
      v16bf a  = frag_ld(pa + k0);
      v16bf b0 = frag_ld(pb0 + k0);
      v16bf b1 = frag_ld(pb1 + k0);
      s0 = wmma_bf16(a, b0, s0);
      s1 = wmma_bf16(a, b1, s1);
    }
    // P = S .* adj ; diagonal: mult==1 -> P = adj. Stage as bf16 16x32 in LDS.
#pragma unroll
    for (int g = 0; g < 8; ++g) {
      int gm = g + lh * 8;
      int gi = i0 + gm;
      const float* arow = adj + (size_t)gi * NV_ + j0;
      float a0 = arow[ln];
      float a1 = arow[16 + ln];
      float v0 = (gi == j0 + ln)      ? a0 : s0[g] * a0;
      float v1 = (gi == j0 + 16 + ln) ? a1 : s1[g] * a1;
      pl[gm * 40 + ln]      = (__bf16)v0;
      pl[gm * 40 + 16 + ln] = (__bf16)v1;
    }
    // Reload P as A-fragment (wave-private LDS; compiler inserts dscnt waits)
    v16bf pfrag = frag_ld(pl + ln * 40 + lh * 8);
#pragma unroll
    for (int t = 0; t < 8; ++t) {
      v16bf vb = frag_ld(Vt + (size_t)(t * 16 + ln) * NV_ + j0 + lh * 8);
      o[t] = wmma_bf16(pfrag, vb, o[t]);
    }
  }
#pragma unroll
  for (int t = 0; t < 8; ++t)
#pragma unroll
    for (int g = 0; g < 8; ++g)
      unsafeAtomicAdd(&Xn[(size_t)(i0 + g + lh * 8) * NHID_ + t * 16 + ln], o[t][g]);
}

// d2[i] = relu(Xn[i,:] + b1) . p2
__global__ __launch_bounds__(256) void k_d2(const float* __restrict__ Xn,
                                            const float* __restrict__ b1,
                                            const float* __restrict__ p2,
                                            float* __restrict__ d2) {
  int i = blockIdx.x * 256 + threadIdx.x;
  float s = 0.f;
  for (int f = 0; f < NHID_; ++f) {
    float x = Xn[i * NHID_ + f] + b1[f];
    x = x > 0.f ? x : 0.f;
    s += x * p2[f];
  }
  d2[i] = s;
}

// Transpose T -> bufB = bf16(T^T) and bufA = bf16(T^T * d2[v])  [NE, NV]
__global__ __launch_bounds__(256) void k_ttr(const float* __restrict__ T,
                                             const float* __restrict__ d2,
                                             __bf16* __restrict__ TtrRaw,
                                             __bf16* __restrict__ TtrS2) {
  __shared__ float tile[32][33];
  int e0 = blockIdx.x * 32;
  int v0 = blockIdx.y * 32;
  int tx = threadIdx.x & 31;
  int ty = threadIdx.x >> 5;  // 0..7
  for (int r = ty; r < 32; r += 8)
    tile[r][tx] = T[(size_t)(v0 + r) * NE_ + e0 + tx];
  __syncthreads();
  float dv = d2[v0 + tx];
  for (int r = ty; r < 32; r += 8) {
    float val = tile[tx][r];  // = T[v0+tx, e0+r]
    size_t oidx = (size_t)(e0 + r) * NV_ + v0 + tx;
    TtrRaw[oidx] = (__bf16)val;
    TtrS2[oidx]  = (__bf16)(val * dv);
  }
}

// ---- stage 2: Zout += (S_e .* adj_e, diag fixed) @ HeW2 ---------------------
// About 2/3 of total FLOPs. Each wave owns TWO 16-row strips and shares the B and Vt
// fragments between them: 4 fragment loads -> 4 WMMAs (32 B/WMMA of L2 traffic
// instead of 48), and total B-side L2 traffic halves.
__global__ __launch_bounds__(256) void k_stage2(
    const __bf16* __restrict__ Ts,   // [NE, NV]  T^T * d2 (bf16)
    const __bf16* __restrict__ Tr,   // [NE, NV]  T^T      (bf16)
    const __bf16* __restrict__ Vt,   // [16, NE]  (relu(Z)@W2)^T padded (bf16)
    const float*  __restrict__ adj,  // [NE, NE]
    float* __restrict__ Zo)          // [NE, NCLS] (atomic accumulate)
{
  __shared__ __align__(16) __bf16 pst[8][2][16 * 40];
  const int lane = threadIdx.x & 31;
  const int wave = threadIdx.x >> 5;
  const int ln = lane & 15;
  const int lh = lane >> 4;
  const int i0 = (blockIdx.x * 8 + wave) * 32;  // two strips: i0, i0+16
  const int jb = blockIdx.y * (NE_ / 8);

  v8f o0 = (v8f){0, 0, 0, 0, 0, 0, 0, 0};
  v8f o1 = (v8f){0, 0, 0, 0, 0, 0, 0, 0};
  const __bf16* pa0 = Ts + (size_t)(i0 + ln) * NV_ + lh * 8;
  const __bf16* pa1 = Ts + (size_t)(i0 + 16 + ln) * NV_ + lh * 8;

  for (int j0 = jb; j0 < jb + NE_ / 8; j0 += 32) {
    v8f s00 = (v8f){0, 0, 0, 0, 0, 0, 0, 0};
    v8f s01 = (v8f){0, 0, 0, 0, 0, 0, 0, 0};
    v8f s10 = (v8f){0, 0, 0, 0, 0, 0, 0, 0};
    v8f s11 = (v8f){0, 0, 0, 0, 0, 0, 0, 0};
    const __bf16* pb0 = Tr + (size_t)(j0 + ln) * NV_ + lh * 8;
    const __bf16* pb1 = Tr + (size_t)(j0 + 16 + ln) * NV_ + lh * 8;
    for (int k0 = 0; k0 < NV_; k0 += 32) {
      v16bf a0 = frag_ld(pa0 + k0);
      v16bf a1 = frag_ld(pa1 + k0);
      v16bf b0 = frag_ld(pb0 + k0);
      v16bf b1 = frag_ld(pb1 + k0);
      s00 = wmma_bf16(a0, b0, s00);
      s01 = wmma_bf16(a0, b1, s01);
      s10 = wmma_bf16(a1, b0, s10);
      s11 = wmma_bf16(a1, b1, s11);
    }
    // Vt fragment shared by both strips for this j-pair
    v16bf vb = frag_ld(Vt + (size_t)ln * NE_ + j0 + lh * 8);
#pragma unroll
    for (int strip = 0; strip < 2; ++strip) {
      __bf16* pl = pst[wave][strip];
      v8f sa = strip ? s10 : s00;
      v8f sb = strip ? s11 : s01;
      int ibase = i0 + strip * 16;
#pragma unroll
      for (int g = 0; g < 8; ++g) {
        int gm = g + lh * 8;
        int gi = ibase + gm;
        const float* arow = adj + (size_t)gi * NE_ + j0;
        float a0 = arow[ln];
        float a1 = arow[16 + ln];
        float v0 = (gi == j0 + ln)      ? a0 : sa[g] * a0;
        float v1 = (gi == j0 + 16 + ln) ? a1 : sb[g] * a1;
        pl[gm * 40 + ln]      = (__bf16)v0;
        pl[gm * 40 + 16 + ln] = (__bf16)v1;
      }
      v16bf pfrag = frag_ld(pl + ln * 40 + lh * 8);
      if (strip == 0) o0 = wmma_bf16(pfrag, vb, o0);
      else            o1 = wmma_bf16(pfrag, vb, o1);
    }
  }
  if (ln < NCLS_) {
#pragma unroll
    for (int g = 0; g < 8; ++g) {
      unsafeAtomicAdd(&Zo[(size_t)(i0 + g + lh * 8) * NCLS_ + ln], o0[g]);
      unsafeAtomicAdd(&Zo[(size_t)(i0 + 16 + g + lh * 8) * NCLS_ + ln], o1[g]);
    }
  }
}

// ---- log_softmax over axis 0 (per class column); b2 cancels -----------------
__global__ __launch_bounds__(256) void k_lse(const float* __restrict__ Zo,
                                             float* __restrict__ lse) {
  int c = blockIdx.x;
  __shared__ float red[256];
  float mx = -INFINITY;
  for (int r = threadIdx.x; r < NE_; r += 256) mx = fmaxf(mx, Zo[r * NCLS_ + c]);
  red[threadIdx.x] = mx;
  __syncthreads();
  for (int s = 128; s > 0; s >>= 1) {
    if (threadIdx.x < s) red[threadIdx.x] = fmaxf(red[threadIdx.x], red[threadIdx.x + s]);
    __syncthreads();
  }
  mx = red[0];
  __syncthreads();
  float sum = 0.f;
  for (int r = threadIdx.x; r < NE_; r += 256) sum += expf(Zo[r * NCLS_ + c] - mx);
  red[threadIdx.x] = sum;
  __syncthreads();
  for (int s = 128; s > 0; s >>= 1) {
    if (threadIdx.x < s) red[threadIdx.x] += red[threadIdx.x + s];
    __syncthreads();
  }
  if (threadIdx.x == 0) lse[c] = mx + logf(red[0]);
}

__global__ __launch_bounds__(256) void k_out(const float* __restrict__ Zo,
                                             const float* __restrict__ lse,
                                             float* __restrict__ out) {
  int idx = blockIdx.x * 256 + threadIdx.x;
  out[idx] = Zo[idx] - lse[idx & (NCLS_ - 1)];
}

// ---- launch -----------------------------------------------------------------
extern "C" void kernel_launch(void* const* d_in, const int* in_sizes, int n_in,
                              void* d_out, int out_size, void* d_ws, size_t ws_size,
                              hipStream_t stream) {
  (void)in_sizes; (void)n_in; (void)out_size; (void)ws_size;
  const float* X    = (const float*)d_in[0];
  const float* Z    = (const float*)d_in[1];
  const float* adje = (const float*)d_in[2];
  const float* adjv = (const float*)d_in[3];
  const float* T    = (const float*)d_in[4];
  const float* W1   = (const float*)d_in[5];
  const float* b1   = (const float*)d_in[6];
  const float* p1   = (const float*)d_in[7];
  const float* W2   = (const float*)d_in[8];
  // d_in[9] = b2: constant per class column, cancels in log_softmax(axis=0)
  const float* p2   = (const float*)d_in[10];
  float* out = (float*)d_out;

  char* ws = (char*)d_ws;
  size_t off = 0;
  auto carve = [&](size_t bytes) -> void* {
    void* p = ws + off;
    off = (off + bytes + 255) & ~(size_t)255;
    return p;
  };
  __bf16* bufA  = (__bf16*)carve((size_t)NV_ * NE_ * 2);  // T*d1 -> T^T*d2
  __bf16* bufB  = (__bf16*)carve((size_t)NV_ * NE_ * 2);  // T    -> T^T
  __bf16* HvWT  = (__bf16*)carve((size_t)NHID_ * NV_ * 2);
  __bf16* HeW2T = (__bf16*)carve((size_t)16 * NE_ * 2);
  float*  d1    = (float*)carve((size_t)NE_ * 4);
  float*  d2    = (float*)carve((size_t)NV_ * 4);
  float*  Xn    = (float*)carve((size_t)NV_ * NHID_ * 4);
  float*  Zo    = (float*)carve((size_t)NE_ * NCLS_ * 4);
  float*  lse   = (float*)carve(256);

  hipMemsetAsync(Xn, 0, (size_t)NV_ * NHID_ * 4, stream);
  hipMemsetAsync(Zo, 0, (size_t)NE_ * NCLS_ * 4, stream);

  k_d1<<<NE_ / 256, 256, 0, stream>>>(Z, p1, d1);
  k_prep<<<(int)(((size_t)NV_ * NE_) / 256), 256, 0, stream>>>(T, d1, bufA, bufB);
  k_hvwt<<<(NHID_ * NV_) / 256, 256, 0, stream>>>(X, W1, HvWT);
  k_hew2t<<<(16 * NE_) / 256, 256, 0, stream>>>(Z, W2, HeW2T);

  k_stage1<<<dim3(NV_ / 16 / 8, 8), 256, 0, stream>>>(bufA, bufB, HvWT, adjv, Xn);
  k_d2<<<NV_ / 256, 256, 0, stream>>>(Xn, b1, p2, d2);
  k_ttr<<<dim3(NE_ / 32, NV_ / 32), 256, 0, stream>>>(T, d2, bufB, bufA);
  k_stage2<<<dim3(NE_ / 32 / 8, 8), 256, 0, stream>>>(bufA, bufB, HeW2T, adje, Zo);

  k_lse<<<NCLS_, 256, 0, stream>>>(Zo, lse);
  k_out<<<(NE_ * NCLS_) / 256, 256, 0, stream>>>(Zo, lse, out);
}
